// RNNTransducer_51599737094226
// MI455X (gfx1250) — matrix operand order repaired and checked by reading, not compiled, not run on hardware
//
#include <hip/hip_runtime.h>
#include <hip/hip_bf16.h>
#include <math.h>

// ---- problem constants (match reference) ----
#define B_   16
#define T_   512
#define U_   128
#define F_   80
#define FP_  96                // F padded to a multiple of 32 (zero-filled)
#define H_   1024
#define P_   512
#define V_   4096
#define NENC 4
#define GH   (4 * H_)          // 4096 gate width

typedef __bf16 bf16_t;
typedef bf16_t v16bf __attribute__((ext_vector_type(16)));
typedef bf16_t v8bf  __attribute__((ext_vector_type(8)));
typedef float  v8f   __attribute__((ext_vector_type(8)));

union ABu { v16bf v; v8bf h[2]; };

__device__ __forceinline__ float sigf(float x) { return 1.0f / (1.0f + __expf(-x)); }

// =====================================================================
// Generic bf16 WMMA GEMM:  out = act( A[M,K] * W + bias + Cadd ),
// W supplied pre-transposed as Wt[N,K] bf16.  One wave = one 16x16 tile.
// K MUST be a multiple of 32 (96/512/1024 here) -> branch-free inner loop:
//   2 b128 loads for A + 2 b128 loads for B + 1 v_wmma_f32_16x16x32_bf16.
// =====================================================================
__global__ void wmma_gemm_bf16(const bf16_t* __restrict__ A,
                               const bf16_t* __restrict__ Wt,
                               const float*  __restrict__ bias,    // [N] or null
                               const float*  __restrict__ Cadd,    // [M,N] or null
                               float*        __restrict__ outF,    // fp32 out (ld=ldOutF) or null
                               bf16_t*       __restrict__ outB,    // bf16 out (ld=N) or null
                               const int*    __restrict__ rowmask, // per-row active or null
                               int M, int N, int K, int ldOutF, int actTanh)
{
  const int wave = threadIdx.x >> 5;
  const int lane = threadIdx.x & 31;
  const int half = lane >> 4;      // 0: lanes 0-15, 1: lanes 16-31
  const int mr   = lane & 15;

  const int nt = blockIdx.x * (blockDim.x >> 5) + wave;   // N-tile index
  const int n0 = nt * 16;
  if (n0 >= N) return;
  const int m0 = blockIdx.y * 16;

  // ISA 7.12.2 fragment layouts:
  //  A 16x32 bf16: lane (half,mr) holds two contiguous 8-elem K runs at
  //                half*8 and 16+half*8 of row M=mr.
  //  B 32x16:      lane (half,mr) holds two contiguous 8-elem K runs at
  //                half*8 and 16+half*8 of column N=mr  (Wt is [N,K], so row runs).
  const bf16_t* Ap = A  + (size_t)(m0 + mr) * K + half * 8;
  const bf16_t* Bp = Wt + (size_t)(n0 + mr) * K + half * 8;

  v8f acc = {};
#pragma unroll 4
  for (int k0 = 0; k0 < K; k0 += 32) {
    ABu a, b;
    a.h[0] = *(const v8bf*)(Ap + k0);
    a.h[1] = *(const v8bf*)(Ap + k0 + 16);
    b.h[0] = *(const v8bf*)(Bp + k0);
    b.h[1] = *(const v8bf*)(Bp + k0 + 16);
    acc = __builtin_amdgcn_wmma_f32_16x16x32_bf16(false, a.v, false, b.v,
                                                  (short)0, acc, false, false);
  }

  const int n  = n0 + mr;
  const float bn = bias ? bias[n] : 0.0f;
#pragma unroll
  for (int r = 0; r < 8; ++r) {
    const int m = m0 + half * 8 + r;          // C/D layout: VGPR r -> M = r + 8*half
    float v = acc[r] + bn;
    if (Cadd) v += Cadd[(size_t)m * N + n];
    if (actTanh) v = tanhf(v);
    if (rowmask && !rowmask[m]) v = 0.0f;
    if (outF) outF[(size_t)m * ldOutF + n] = v;
    if (outB) outB[(size_t)m * N + n] = (bf16_t)v;
  }
}

// ---- LSTM pointwise: gates[rows,4H] (i,j,f,o) + c -> c', m = sig(o)*tanh(c') ----
__global__ void k_lstm_pw(const float* __restrict__ g, const float* __restrict__ cin,
                          float* __restrict__ cout, bf16_t* __restrict__ m, int rows)
{
  const int idx = blockIdx.x * blockDim.x + threadIdx.x;
  if (idx >= rows * H_) return;
  const int r = idx / H_, h = idx % H_;
  const float* gr = g + (size_t)r * GH;
  const float i = gr[h], j = gr[H_ + h], f = gr[2 * H_ + h], o = gr[3 * H_ + h];
  const float c  = cin ? cin[idx] : 0.0f;
  const float c2 = sigf(f + 1.0f) * c + sigf(i) * tanhf(j);   // forget bias 1.0
  if (cout) cout[idx] = c2;
  m[idx] = (bf16_t)(sigf(o) * tanhf(c2));
}

// ---- decoder layer-0 for every possible label v (scalar input, zero state) ----
__global__ void k_dec0(const float* __restrict__ Wx0, const float* __restrict__ b0,
                       bf16_t* __restrict__ m0)
{
  const int idx = blockIdx.x * blockDim.x + threadIdx.x;
  if (idx >= V_ * H_) return;
  const int v = idx / H_, h = idx % H_;
  const float xv = (float)v;
  const float i = xv * Wx0[h]          + b0[h];
  const float j = xv * Wx0[H_ + h]     + b0[H_ + h];
  const float o = xv * Wx0[3 * H_ + h] + b0[3 * H_ + h];
  const float c2 = sigf(i) * tanhf(j);            // c=0 => forget term vanishes
  m0[idx] = (bf16_t)(sigf(o) * tanhf(c2));
}

// ---- greedy-decode init: state=0, s0 = enc[:,0] + G[0] ----
__global__ void k_dec_init(const bf16_t* __restrict__ enc, const float* __restrict__ G,
                           int* st_t, int* st_u, int* st_y, int* act,
                           bf16_t* __restrict__ s)
{
  const int idx = blockIdx.x * blockDim.x + threadIdx.x;
  if (idx >= B_ * P_) return;
  const int b = idx / P_, p = idx % P_;
  s[idx] = (bf16_t)((float)enc[(size_t)b * P_ + p] + G[p]);
  if (idx < B_) { st_t[idx] = 0; st_u[idx] = 0; st_y[idx] = 0; act[idx] = 1; }
}

// ---- per-step update: wave b = argmax over V, state update, prepare s_next ----
__global__ void k_dec_update(const float* __restrict__ logits, int ldl,
                             const bf16_t* __restrict__ enc, const float* __restrict__ G,
                             int* st_t, int* st_u, int* st_y, int* act,
                             int* labels, bf16_t* __restrict__ s)
{
  const int b = threadIdx.x >> 5;
  const int lane = threadIdx.x & 31;
  const float* row = logits + (size_t)b * ldl;

  float best = -3.4e38f; int bi = 0;
  for (int k = lane; k < V_; k += 32) {        // strict > keeps first max per lane
    const float v = row[k];
    if (v > best) { best = v; bi = k; }
  }
#pragma unroll
  for (int off = 16; off >= 1; off >>= 1) {    // wave32 reduce, smaller idx on tie
    const float ob = __shfl_xor(best, off, 32);
    const int   oi = __shfl_xor(bi,   off, 32);
    if (ob > best || (ob == best && oi < bi)) { best = ob; bi = oi; }
  }

  const int tb = st_t[b], ub = st_u[b], yb = st_y[b];
  const bool a    = (tb < T_) && (ub < U_);
  const bool emit = a && (bi != 0);
  const int t2 = tb + (a ? 1 : 0);
  const int u2 = ub + (emit ? 1 : 0);
  const int y2 = emit ? bi : yb;
  if (lane == 0) {
    if (emit) labels[b * U_ + ub] = bi;
    st_t[b] = t2; st_u[b] = u2; st_y[b] = y2;
    act[b] = (t2 < T_ && u2 < U_) ? 1 : 0;
  }
  const int tc = (t2 < T_) ? t2 : T_ - 1;
  for (int p = lane; p < P_; p += 32)
    s[b * P_ + p] = (bf16_t)((float)enc[(size_t)(tc * B_ + b) * P_ + p] +
                             G[(size_t)y2 * P_ + p]);
}

// ---- utilities ----
__global__ void k_fill_u32(unsigned int* p, unsigned int v, int n)
{ const int i = blockIdx.x * blockDim.x + threadIdx.x; if (i < n) p[i] = v; }

// Wt[N,Kpad] = bf16(W[K,N]^T), zero-filled for k in [K,Kpad)
__global__ void k_wt_bf16(const float* __restrict__ W, bf16_t* __restrict__ Wt,
                          int K, int Kpad, int N)
{
  const int idx = blockIdx.x * blockDim.x + threadIdx.x;
  if (idx >= N * Kpad) return;
  const int n = idx / Kpad, k = idx % Kpad;
  Wt[idx] = (k < K) ? (bf16_t)W[(size_t)k * N + n] : (bf16_t)0.0f;
}

__global__ void k_x_bf16(const float* __restrict__ x, bf16_t* __restrict__ o)
{ // [B,T,F] -> [T*B, FP] bf16, zero-padded features
  const int idx = blockIdx.x * blockDim.x + threadIdx.x;
  if (idx >= T_ * B_ * FP_) return;
  const int t = idx / (B_ * FP_), r = idx % (B_ * FP_);
  const int b = r / FP_, f = r % FP_;
  o[idx] = (f < F_) ? (bf16_t)x[(size_t)b * T_ * F_ + (size_t)t * F_ + f]
                    : (bf16_t)0.0f;
}

__global__ void k_final(const int* __restrict__ labels, const int* __restrict__ u,
                        float* __restrict__ out)
{
  const int i = blockIdx.x * blockDim.x + threadIdx.x;
  if (i < B_ * U_)            out[i] = (float)labels[i];
  else if (i < B_ * U_ + B_)  out[i] = (float)u[i - B_ * U_];
}

// =====================================================================
static inline void gemm(hipStream_t s, const bf16_t* A, const bf16_t* Wt,
                        const float* bias, const float* Cadd, float* outF,
                        bf16_t* outB, const int* rowmask,
                        int M, int N, int K, int ldOutF, int actTanh)
{
  dim3 grid((N / 16 + 7) / 8, M / 16);
  wmma_gemm_bf16<<<grid, 256, 0, s>>>(A, Wt, bias, Cadd, outF, outB, rowmask,
                                      M, N, K, ldOutF, actTanh);
}
static inline int cdiv(int a, int b) { return (a + b - 1) / b; }

extern "C" void kernel_launch(void* const* d_in, const int* in_sizes, int n_in,
                              void* d_out, int out_size, void* d_ws, size_t ws_size,
                              hipStream_t stream)
{
  (void)in_sizes; (void)n_in; (void)out_size; (void)ws_size;
  const float* x    = (const float*)d_in[0];
  const float* eWx0 = (const float*)d_in[1];
  const float* eWxR = (const float*)d_in[2];
  const float* eWh  = (const float*)d_in[3];
  const float* eB   = (const float*)d_in[4];
  const float* eWp  = (const float*)d_in[5];
  const float* dWx0 = (const float*)d_in[6];
  const float* dWx1 = (const float*)d_in[7];
  /* d_in[8] dec_Wh: unused — decoder h-state is identically zero in reference */
  const float* dB   = (const float*)d_in[9];
  const float* dWp  = (const float*)d_in[10];
  const float* Wj   = (const float*)d_in[11];
  const float* bj   = (const float*)d_in[12];

  // ---- carve scratch (256B aligned bump allocator) ----
  char* w = (char*)d_ws;
  auto carve = [&](size_t bytes) -> char* {
    char* p = w; w += (bytes + 255) & ~(size_t)255; return p;
  };
  bf16_t* xbf  = (bf16_t*)carve((size_t)T_ * B_ * FP_ * 2);
  bf16_t* Wx0T = (bf16_t*)carve((size_t)GH * FP_ * 2);
  bf16_t* WxT[NENC]; WxT[0] = nullptr;
  for (int l = 1; l < NENC; ++l) WxT[l] = (bf16_t*)carve((size_t)GH * P_ * 2);
  bf16_t* WhT[NENC];
  for (int l = 0; l < NENC; ++l) WhT[l] = (bf16_t*)carve((size_t)GH * P_ * 2);
  bf16_t* WpT[NENC];
  for (int l = 0; l < NENC; ++l) WpT[l] = (bf16_t*)carve((size_t)P_ * H_ * 2);
  bf16_t* dWpT0  = (bf16_t*)carve((size_t)P_ * H_ * 2);
  bf16_t* dWpT1  = (bf16_t*)carve((size_t)P_ * H_ * 2);
  bf16_t* dWx1T  = (bf16_t*)carve((size_t)GH * P_ * 2);
  bf16_t* WjT    = (bf16_t*)carve((size_t)V_ * P_ * 2);
  float*  gatesAll = (float*)carve((size_t)T_ * B_ * GH * 4);  // reused as dec gates1
  bf16_t* houtA  = (bf16_t*)carve((size_t)T_ * B_ * P_ * 2);
  bf16_t* houtB  = (bf16_t*)carve((size_t)T_ * B_ * P_ * 2);
  float*  gatesBuf = (float*)carve((size_t)B_ * GH * 4);
  float*  cBuf  = (float*)carve((size_t)B_ * H_ * 4);
  bf16_t* mBuf  = (bf16_t*)carve((size_t)B_ * H_ * 2);
  bf16_t* m0    = (bf16_t*)carve((size_t)V_ * H_ * 2);          // reused as m1
  bf16_t* H1    = (bf16_t*)carve((size_t)V_ * P_ * 2);
  float*  Gtab  = (float*)carve((size_t)V_ * P_ * 4);
  bf16_t* sBuf  = (bf16_t*)carve((size_t)B_ * P_ * 2);
  int* st_t   = (int*)carve(B_ * 4);
  int* st_u   = (int*)carve(B_ * 4);
  int* st_y   = (int*)carve(B_ * 4);
  int* actF   = (int*)carve(B_ * 4);
  int* labels = (int*)carve((size_t)B_ * U_ * 4);

  // ---- one-time (per call) weight transpose+convert to bf16 [N,K] ----
  auto wt = [&](const float* W, bf16_t* Wt, int K, int Kpad, int N) {
    k_wt_bf16<<<cdiv(N * Kpad, 256), 256, 0, stream>>>(W, Wt, K, Kpad, N);
  };
  wt(eWx0, Wx0T, F_, FP_, GH);
  for (int l = 1; l < NENC; ++l) wt(eWxR + (size_t)(l - 1) * P_ * GH, WxT[l], P_, P_, GH);
  for (int l = 0; l < NENC; ++l) wt(eWh  + (size_t)l * P_ * GH, WhT[l], P_, P_, GH);
  for (int l = 0; l < NENC; ++l) wt(eWp  + (size_t)l * H_ * P_, WpT[l], H_, H_, P_);
  wt(dWp,                   dWpT0, H_, H_, P_);
  wt(dWp + (size_t)H_ * P_, dWpT1, H_, H_, P_);
  wt(dWx1, dWx1T, P_, P_, GH);
  wt(Wj,   WjT,   P_, P_, GH);
  k_x_bf16<<<cdiv(T_ * B_ * FP_, 256), 256, 0, stream>>>(x, xbf);

  // ================= encoder: 4 stacked LSTMP layers =================
  const bf16_t* inBuf = xbf;
  int Kin = FP_;
  bf16_t* outBuf = houtA;
  for (int l = 0; l < NENC; ++l) {
    // all-timestep input projection (parallel): gatesAll = inBuf @ Wx + b
    gemm(stream, inBuf, (l == 0 ? Wx0T : WxT[l]), eB + (size_t)l * GH,
         nullptr, gatesAll, nullptr, nullptr, T_ * B_, GH, Kin, GH, 0);
    k_fill_u32<<<cdiv(B_ * H_, 256), 256, 0, stream>>>((unsigned*)cBuf, 0u, B_ * H_);
    for (int t = 0; t < T_; ++t) {
      const float* g = gatesAll + (size_t)t * B_ * GH;
      if (t > 0) {   // serial part: gates += h_{t-1} @ Wh
        gemm(stream, outBuf + (size_t)(t - 1) * B_ * P_, WhT[l], nullptr, g,
             gatesBuf, nullptr, nullptr, B_, GH, P_, GH, 0);
        g = gatesBuf;
      }
      k_lstm_pw<<<cdiv(B_ * H_, 256), 256, 0, stream>>>(g, cBuf, cBuf, mBuf, B_);
      gemm(stream, mBuf, WpT[l], nullptr, nullptr, nullptr,
           outBuf + (size_t)t * B_ * P_, nullptr, B_, P_, H_, 0, 0);
    }
    inBuf = outBuf; Kin = P_;
    outBuf = (outBuf == houtA) ? houtB : houtA;
  }
  const bf16_t* enc = inBuf;   // [T*B, P] bf16 encoder output

  // ===== decoder g-table: G[v] = dec_step(v) for all labels (parallel) =====
  k_dec0<<<cdiv(V_ * H_, 256), 256, 0, stream>>>(dWx0, dB, m0);
  gemm(stream, m0, dWpT0, nullptr, nullptr, nullptr, H1, nullptr, V_, P_, H_, 0, 0);
  gemm(stream, H1, dWx1T, dB + GH, nullptr, gatesAll, nullptr, nullptr, V_, GH, P_, GH, 0);
  k_lstm_pw<<<cdiv(V_ * H_, 256), 256, 0, stream>>>(gatesAll, nullptr, nullptr, m0, V_);
  gemm(stream, m0, dWpT1, nullptr, nullptr, Gtab, nullptr, nullptr, V_, P_, H_, P_, 0);

  // ================= greedy RNN-T decode (serial, T+U steps) =================
  k_fill_u32<<<cdiv(B_ * U_, 256), 256, 0, stream>>>((unsigned*)labels, 0u, B_ * U_);
  float* logitsBase = (float*)d_out + B_ * U_ + B_;   // [B, T+U, V]
  const int ldl = (T_ + U_) * V_;
  k_dec_init<<<cdiv(B_ * P_, 256), 256, 0, stream>>>(enc, Gtab, st_t, st_u, st_y,
                                                     actF, sBuf);
  for (int s = 0; s < T_ + U_; ++s) {
    // joint: tanh((f+g)·Wj + bj), inactive rows zeroed, written straight to d_out
    gemm(stream, sBuf, WjT, bj, nullptr, logitsBase + (size_t)s * V_, nullptr,
         actF, B_, V_, P_, ldl, 1);
    k_dec_update<<<1, 512, 0, stream>>>(logitsBase + (size_t)s * V_, ldl, enc, Gtab,
                                        st_t, st_u, st_y, actF, labels, sBuf);
  }
  k_final<<<cdiv(B_ * U_ + B_, 256), 256, 0, stream>>>(labels, st_u, (float*)d_out);
}
